// Transformer_81149112091207
// MI455X (gfx1250) — matrix operand order repaired and checked by reading, not compiled, not run on hardware
//
#include <hip/hip_runtime.h>
#include <hip/hip_bf16.h>
#include <math.h>

// ---------------------------------------------------------------------------
// Multimodal transformer forward (B=8, S=512, E=768, H=24, DH=32, L=16, FF=6144)
// HBM-bound on fp32 weight streaming (~700MB/call @ 23.3TB/s).  fp32 storage,
// bf16 in-register for v_wmma_f32_16x16x32_bf16 with f32 accumulation.
// All GEMM call sites guarantee M % 64 == 0 and K % 32 == 0; only N may be
// partial (N=32 for QKV / P*V), so tile staging is branch-free except one
// hoisted per-thread N guard.
// ---------------------------------------------------------------------------

#define B_    8
#define T1_   256
#define GRID_ 16
#define E_    768
#define H_    24
#define DH_   32
#define L_    16
#define FF_   6144
#define T2_   (GRID_ * GRID_)
#define S_    (T1_ + T2_)
#define EPS_  1.1920929e-07f

typedef __bf16 bf16_t;
typedef bf16_t bf16x16 __attribute__((ext_vector_type(16)));
typedef bf16_t bf16x8  __attribute__((ext_vector_type(8)));
typedef float  f32x8   __attribute__((ext_vector_type(8)));

union FragAB { bf16x16 v; bf16x8 h[2]; };

__device__ __forceinline__ float gelu_exact(float x) {
    return 0.5f * x * (1.0f + erff(x * 0.70710678118654752f));
}

__device__ __forceinline__ bf16x8 pack8(float4 a, float4 b) {
    bf16x8 p;
    p[0] = (bf16_t)a.x; p[1] = (bf16_t)a.y; p[2] = (bf16_t)a.z; p[3] = (bf16_t)a.w;
    p[4] = (bf16_t)b.x; p[5] = (bf16_t)b.y; p[6] = (bf16_t)b.z; p[7] = (bf16_t)b.w;
    return p;
}

// ---------------------------------------------------------------------------
// Generic batched GEMM:  C[m,n] = act( scale * sum_k A[m,k]*B[n,k] + bias[n] )
//   A: M x K row-major (lda), per-batch strides sA0/sA1 (batch = i0*bd1 + i1)
//   B: stored N x K row-major (transB=0) or K x N row-major (transB=1)
//   Tiles 64x64x32; 256 threads = 8 waves; wave = 16(M) x 32(N) = 2 wmma frags.
//   REQUIRES: M % 64 == 0, K % 32 == 0, A/B rows 16B-aligned (true at all call
//   sites here).
// ---------------------------------------------------------------------------
__global__ __launch_bounds__(256) void gemm_bf16_wmma(
    const float* __restrict__ A, long lda, long sA0, long sA1,
    const float* __restrict__ Bw, long ldb, long sB0, long sB1,
    const float* __restrict__ bias, long sb0, long sb1,
    float* __restrict__ C, long ldc, long sC0, long sC1,
    int M, int N, int K, int bd1, float scale, int act, int transB)
{
    __shared__ __attribute__((aligned(16))) bf16_t As[64][40];  // +8 pad: conflict-free b128 reads
    __shared__ __attribute__((aligned(16))) bf16_t Bs[64][40];

    const int z  = blockIdx.z;
    const int i1 = z % bd1;
    const int i0 = z / bd1;
    A  += (long)i0 * sA0 + (long)i1 * sA1;
    Bw += (long)i0 * sB0 + (long)i1 * sB1;
    if (bias) bias += (long)i0 * sb0 + (long)i1 * sb1;
    C  += (long)i0 * sC0 + (long)i1 * sC1;

    const int M0   = blockIdx.y * 64;
    const int N0   = blockIdx.x * 64;
    const int t    = threadIdx.x;
    const int wave = t >> 5;
    const int lane = t & 31;
    const int wm   = (wave & 3) * 16;   // wave's M offset in tile
    const int wn   = (wave >> 2) * 32;  // wave's N offset in tile

    // staging ownership (branch-free mappings, guards loop-invariant)
    const int sr  = t >> 2;             // 0..63  : tile row (A: m, B: n)
    const int sc  = (t & 3) * 8;        // 0,8,16,24 : k chunk start
    const int tn  = t & 63;             // transB: tile n column (loop-invariant)
    const int tcb = (t >> 6) * 8;       // transB: k chunk start
    const bool bRowValid = (N0 + sr) < N;   // non-trans B guard
    const bool bColValid = (N0 + tn) < N;   // transB   B guard

    f32x8 acc0 = {0.f,0.f,0.f,0.f,0.f,0.f,0.f,0.f};
    f32x8 acc1 = {0.f,0.f,0.f,0.f,0.f,0.f,0.f,0.f};

    for (int k0 = 0; k0 < K; k0 += 32) {
        // ---- stage A tile (64x32): 2x float4 -> bf16x8 -> one ds_store_b128 ----
        {
            const float* ap = A + (long)(M0 + sr) * lda + (k0 + sc);
            float4 fa = *(const float4*)ap;
            float4 fb = *(const float4*)(ap + 4);
            *(bf16x8*)&As[sr][sc] = pack8(fa, fb);
        }
        // ---- stage B tile as [N][K] ----
        if (!transB) {
            bf16x8 pk;
            if (bRowValid) {
                const float* bp = Bw + (long)(N0 + sr) * ldb + (k0 + sc);
                float4 fa = *(const float4*)bp;
                float4 fb = *(const float4*)(bp + 4);
                pk = pack8(fa, fb);
            } else {
                #pragma unroll
                for (int j = 0; j < 8; ++j) pk[j] = (bf16_t)0.0f;
            }
            *(bf16x8*)&Bs[sr][sc] = pk;
        } else {
            bf16x8 pk;
            if (bColValid) {
                const float* bp = Bw + (long)(k0 + tcb) * ldb + (N0 + tn);
                #pragma unroll
                for (int j = 0; j < 8; ++j) pk[j] = (bf16_t)bp[(long)j * ldb];
            } else {
                #pragma unroll
                for (int j = 0; j < 8; ++j) pk[j] = (bf16_t)0.0f;
            }
            *(bf16x8*)&Bs[tn][tcb] = pk;
        }
        __syncthreads();

        // hint next tiles into cache (gfx1250 global_prefetch_b8 path);
        // B (weights) is the dominant HBM stream.
        if (k0 + 32 < K) {
            __builtin_prefetch(&A[(long)(M0 + sr) * lda + (k0 + 32)], 0, 1);
            if (!transB && bRowValid)
                __builtin_prefetch(&Bw[(long)(N0 + sr) * ldb + (k0 + 32)], 0, 1);
        }

        // ---- build fragments per ISA VGPR layouts ----
        // A (16-bit 16x32): lane half h gets K chunks {8h..8h+7, 16+8h..16+8h+7}
        const int mrow = wm + (lane & 15);
        const int ka   = (lane >> 4) * 8;
        FragAB a;
        a.h[0] = *(const bf16x8*)&As[mrow][ka];
        a.h[1] = *(const bf16x8*)&As[mrow][16 + ka];
        // B (16-bit 32x16): lane half h gets K chunks {8h..8h+7, 16+8h..16+8h+7}
        const int nrow = wn + (lane & 15);
        const int kb   = (lane >> 4) * 8;
        FragAB b0, b1;
        b0.h[0] = *(const bf16x8*)&Bs[nrow][kb];
        b0.h[1] = *(const bf16x8*)&Bs[nrow][kb + 16];
        b1.h[0] = *(const bf16x8*)&Bs[nrow + 16][kb];
        b1.h[1] = *(const bf16x8*)&Bs[nrow + 16][kb + 16];

        acc0 = __builtin_amdgcn_wmma_f32_16x16x32_bf16(false, a.v, false, b0.v,
                                                       (short)0, acc0, false, false);
        acc1 = __builtin_amdgcn_wmma_f32_16x16x32_bf16(false, a.v, false, b1.v,
                                                       (short)0, acc1, false, false);
        __syncthreads();
    }

    // ---- epilogue: C/D layout m = r + (lane>>4)*8, n = lane&15 ----
    #pragma unroll
    for (int r = 0; r < 8; ++r) {
        int m = M0 + wm + r + (lane >> 4) * 8;
        #pragma unroll
        for (int j = 0; j < 2; ++j) {
            int n = N0 + wn + j * 16 + (lane & 15);
            if (n >= N) continue;
            float vv = (j ? acc1[r] : acc0[r]) * scale;
            if (bias) vv += bias[n];
            if (act == 1) vv = gelu_exact(vv);
            C[(long)m * ldc + n] = vv;
        }
    }
}

// ---------------------------------------------------------------------------
// Masked softmax over one S_=512 row per wave.  Replicates reference quirk:
// masked entries become -1e-6 and still participate in the softmax.
// rows = B*H*S, grid = rows/8 blocks of 256 (8 waves).
// ---------------------------------------------------------------------------
__global__ __launch_bounds__(256) void softmax_mask_kernel(float* __restrict__ scores)
{
    const int gwave = blockIdx.x * 8 + (threadIdx.x >> 5);
    const int lane  = threadIdx.x & 31;
    const int srow  = gwave % S_;
    float* row = scores + (long)gwave * S_;

    float v[16];
    float mx = -3.0e38f;
    #pragma unroll
    for (int i = 0; i < 16; ++i) {
        int tcol = lane + i * 32;
        float xv = row[tcol];
        bool keep = (srow < T1_) || (tcol < T1_) || (tcol <= srow);
        xv = keep ? xv : -1e-6f;
        v[i] = xv;
        mx = fmaxf(mx, xv);
    }
    #pragma unroll
    for (int off = 16; off >= 1; off >>= 1) mx = fmaxf(mx, __shfl_xor(mx, off, 32));
    float sum = 0.f;
    #pragma unroll
    for (int i = 0; i < 16; ++i) { v[i] = expf(v[i] - mx); sum += v[i]; }
    #pragma unroll
    for (int off = 16; off >= 1; off >>= 1) sum += __shfl_xor(sum, off, 32);
    float inv = 1.0f / sum;
    #pragma unroll
    for (int i = 0; i < 16; ++i) row[lane + i * 32] = v[i] * inv;
}

// ---------------------------------------------------------------------------
// out[row] = rmsnorm(xin[row] (+ yin[row]), w) ; one 256-thread block per row.
// ---------------------------------------------------------------------------
__global__ __launch_bounds__(256) void add_rmsnorm_kernel(
    const float* __restrict__ xin, const float* __restrict__ yin,
    const float* __restrict__ w, float* __restrict__ out)
{
    __shared__ float red[8];
    const long row = blockIdx.x;
    const float* xr = xin + row * E_;
    const float* yr = yin ? (yin + row * E_) : nullptr;
    float v[3];
    float ss = 0.f;
    #pragma unroll
    for (int i = 0; i < 3; ++i) {
        int e = threadIdx.x + i * 256;
        float tv = xr[e] + (yr ? yr[e] : 0.f);
        v[i] = tv;
        ss += tv * tv;
    }
    #pragma unroll
    for (int off = 16; off >= 1; off >>= 1) ss += __shfl_xor(ss, off, 32);
    if ((threadIdx.x & 31) == 0) red[threadIdx.x >> 5] = ss;
    __syncthreads();
    float tot = 0.f;
    #pragma unroll
    for (int i = 0; i < 8; ++i) tot += red[i];
    float rs = rsqrtf(tot * (1.0f / E_) + EPS_);
    #pragma unroll
    for (int i = 0; i < 3; ++i) {
        int e = threadIdx.x + i * 256;
        out[row * E_ + e] = v[i] * rs * w[e];
    }
}

// ---------------------------------------------------------------------------
// Direct conv 4x4, stride 2, pad 1, + bias + relu (NCHW / OIHW).
// ---------------------------------------------------------------------------
__global__ __launch_bounds__(256) void conv4x4s2_relu_kernel(
    const float* __restrict__ in, const float* __restrict__ w,
    const float* __restrict__ bias, float* __restrict__ out,
    int Bn, int Cin, int Hin, int Win, int Cout, int Hout, int Wout)
{
    long idx = (long)blockIdx.x * 256 + threadIdx.x;
    long total = (long)Bn * Cout * Hout * Wout;
    if (idx >= total) return;
    int wo = (int)(idx % Wout);
    int ho = (int)((idx / Wout) % Hout);
    int co = (int)((idx / ((long)Wout * Hout)) % Cout);
    int bb = (int)(idx / ((long)Wout * Hout * Cout));
    float acc = bias[co];
    for (int ci = 0; ci < Cin; ++ci) {
        const float* ip = in + ((long)(bb * Cin + ci) * Hin) * Win;
        const float* wp = w + (long)(co * Cin + ci) * 16;
        #pragma unroll
        for (int ky = 0; ky < 4; ++ky) {
            int iy = ho * 2 + ky - 1;
            if (iy < 0 || iy >= Hin) continue;
            #pragma unroll
            for (int kx = 0; kx < 4; ++kx) {
                int ix = wo * 2 + kx - 1;
                if (ix < 0 || ix >= Win) continue;
                acc += ip[(long)iy * Win + ix] * wp[ky * 4 + kx];
            }
        }
    }
    out[idx] = fmaxf(acc, 0.0f);
}

// x[b, T1+t2, e] = f4[b, e, gy, gx] + row_emb[gy,e] + col_emb[gx,e]
__global__ __launch_bounds__(256) void img_tokens_kernel(
    const float* __restrict__ f4, const float* __restrict__ row_emb,
    const float* __restrict__ col_emb, float* __restrict__ x)
{
    long idx = (long)blockIdx.x * 256 + threadIdx.x;  // over B*T2*E
    int e  = (int)(idx % E_);
    int t2 = (int)((idx / E_) % T2_);
    int b  = (int)(idx / ((long)E_ * T2_));
    int gy = t2 >> 4, gx = t2 & 15;
    float fv = f4[(((long)b * E_ + e) * GRID_ + gy) * GRID_ + gx];
    x[((long)b * S_ + T1_ + t2) * E_ + e] =
        fv + row_emb[(long)gy * E_ + e] + col_emb[(long)gx * E_ + e];
}

// x[b, s, e] = text_emb[text[b,s], e] + text_pos[s, e]
__global__ __launch_bounds__(256) void embed_text_kernel(
    const int* __restrict__ text, const float* __restrict__ temb,
    const float* __restrict__ tpos, float* __restrict__ x)
{
    long idx = (long)blockIdx.x * 256 + threadIdx.x;  // over B*T1*E
    int e = (int)(idx % E_);
    int s = (int)((idx / E_) % T1_);
    int b = (int)(idx / ((long)E_ * T1_));
    int tok = text[(long)b * T1_ + s];
    x[((long)b * S_ + s) * E_ + e] = temb[(long)tok * E_ + e] + tpos[(long)s * E_ + e];
}

// ---------------------------------------------------------------------------
extern "C" void kernel_launch(void* const* d_in, const int* in_sizes, int n_in,
                              void* d_out, int out_size, void* d_ws, size_t ws_size,
                              hipStream_t stream)
{
    (void)in_sizes; (void)n_in; (void)out_size; (void)ws_size;

    const int*   text    = (const int*)  d_in[0];
    const float* images  = (const float*)d_in[1];
    const float* temb    = (const float*)d_in[2];
    const float* tpos    = (const float*)d_in[3];
    const float* cw1 = (const float*)d_in[4];  const float* cb1 = (const float*)d_in[5];
    const float* cw2 = (const float*)d_in[6];  const float* cb2 = (const float*)d_in[7];
    const float* cw3 = (const float*)d_in[8];  const float* cb3 = (const float*)d_in[9];
    const float* cw4 = (const float*)d_in[10]; const float* cb4 = (const float*)d_in[11];
    const float* row_emb = (const float*)d_in[12];
    const float* col_emb = (const float*)d_in[13];
    const float* Wq  = (const float*)d_in[14];
    const float* Wk  = (const float*)d_in[15];
    const float* Wv  = (const float*)d_in[16];
    const float* Wo  = (const float*)d_in[17];
    const float* bo  = (const float*)d_in[18];
    const float* n1w = (const float*)d_in[19];
    const float* n2w = (const float*)d_in[20];
    const float* fw1 = (const float*)d_in[21];
    const float* fb1 = (const float*)d_in[22];
    const float* fw2 = (const float*)d_in[23];
    const float* fb2 = (const float*)d_in[24];
    const float* normw = (const float*)d_in[25];
    const float* lmw   = (const float*)d_in[26];
    const float* lmb   = (const float*)d_in[27];
    float* out = (float*)d_out;

    // workspace layout (floats); total ~276.8MB
    const long XE = (long)B_ * S_ * E_;          // 3,145,728  (== B*H*S*DH)
    float* x    = (float*)d_ws;
    float* tmp1 = x    + XE;                     // attention heads -> (B,S,E)
    float* tmp2 = tmp1 + XE;                     // projection / ffn output
    float* q    = tmp2 + XE;                     // (B,H,S,DH)
    float* kbuf = q    + XE;
    float* vbuf = kbuf + XE;
    float* big  = vbuf + XE;                     // 50,331,648 floats, time-shared
    float* f1 = big;                             // conv features (pre-layer phase)
    float* f2 = f1 + (long)B_ * 32 * 128 * 128;
    float* f3 = f2 + (long)B_ * 64 * 64 * 64;
    float* f4 = f3 + (long)B_ * 128 * 32 * 32;

    auto gemm = [&](const float* Ap, long lda, long sA0, long sA1,
                    const float* Bp, long ldb, long sB0, long sB1,
                    const float* biasp, long sb0, long sb1,
                    float* Cp, long ldc, long sC0, long sC1,
                    int M, int N, int K, int bd0, int bd1,
                    float scale, int act, int transB) {
        dim3 g((unsigned)((N + 63) / 64), (unsigned)((M + 63) / 64), (unsigned)(bd0 * bd1));
        gemm_bf16_wmma<<<g, 256, 0, stream>>>(
            Ap, lda, sA0, sA1, Bp, ldb, sB0, sB1, biasp, sb0, sb1,
            Cp, ldc, sC0, sC1, M, N, K, bd1, scale, act, transB);
    };

    // ---- embeddings + vision stem ----
    embed_text_kernel<<<(B_ * T1_ * E_) / 256, 256, 0, stream>>>(text, temb, tpos, x);
    conv4x4s2_relu_kernel<<<(B_ * 32 * 128 * 128) / 256, 256, 0, stream>>>(
        images, cw1, cb1, f1, B_, 3, 256, 256, 32, 128, 128);
    conv4x4s2_relu_kernel<<<(B_ * 64 * 64 * 64) / 256, 256, 0, stream>>>(
        f1, cw2, cb2, f2, B_, 32, 128, 128, 64, 64, 64);
    conv4x4s2_relu_kernel<<<(B_ * 128 * 32 * 32) / 256, 256, 0, stream>>>(
        f2, cw3, cb3, f3, B_, 64, 64, 64, 128, 32, 32);
    conv4x4s2_relu_kernel<<<(B_ * E_ * 16 * 16) / 256, 256, 0, stream>>>(
        f3, cw4, cb4, f4, B_, 128, 32, 32, E_, 16, 16);
    img_tokens_kernel<<<(B_ * T2_ * E_) / 256, 256, 0, stream>>>(f4, row_emb, col_emb, x);

    const float inv_sqrt_dk = 0.17677669529663687f;  // 1/sqrt(DH)

    for (int l = 0; l < L_; ++l) {
        const float* wq = Wq + (long)l * H_ * DH_ * DH_;
        const float* wk = Wk + (long)l * H_ * DH_ * DH_;
        const float* wv = Wv + (long)l * H_ * DH_ * DH_;

        // QKV: per-(b,h) GEMM S x 32 x 32.  A = x[b, :, h*32:(h+1)*32]
        gemm(x, E_, (long)S_ * E_, DH_,
             wq, DH_, 0, (long)DH_ * DH_,
             nullptr, 0, 0,
             q, DH_, (long)H_ * S_ * DH_, (long)S_ * DH_,
             S_, DH_, DH_, B_, H_, 1.f, 0, 0);
        gemm(x, E_, (long)S_ * E_, DH_,
             wk, DH_, 0, (long)DH_ * DH_,
             nullptr, 0, 0,
             kbuf, DH_, (long)H_ * S_ * DH_, (long)S_ * DH_,
             S_, DH_, DH_, B_, H_, 1.f, 0, 0);
        gemm(x, E_, (long)S_ * E_, DH_,
             wv, DH_, 0, (long)DH_ * DH_,
             nullptr, 0, 0,
             vbuf, DH_, (long)H_ * S_ * DH_, (long)S_ * DH_,
             S_, DH_, DH_, B_, H_, 1.f, 0, 0);

        // scores = (Q K^T) / sqrt(DH)   -> big as (B,H,S,S)
        gemm(q, DH_, (long)H_ * S_ * DH_, (long)S_ * DH_,
             kbuf, DH_, (long)H_ * S_ * DH_, (long)S_ * DH_,
             nullptr, 0, 0,
             big, S_, (long)H_ * S_ * S_, (long)S_ * S_,
             S_, S_, DH_, B_, H_, inv_sqrt_dk, 0, 0);
        softmax_mask_kernel<<<(B_ * H_ * S_) / 8, 256, 0, stream>>>(big);

        // O = P V, heads written interleaved straight into (B,S,E) tmp1
        gemm(big, S_, (long)H_ * S_ * S_, (long)S_ * S_,
             vbuf, DH_, (long)H_ * S_ * DH_, (long)S_ * DH_,
             nullptr, 0, 0,
             tmp1, E_, (long)S_ * E_, DH_,
             S_, DH_, S_, B_, H_, 1.f, 0, 1);

        // attention output projection (+bo), then x = rmsnorm(x + o, n1)
        gemm(tmp1, E_, 0, 0,
             Wo + (long)l * E_ * E_, E_, 0, 0,
             bo + (long)l * E_, 0, 0,
             tmp2, E_, 0, 0,
             B_ * S_, E_, E_, 1, 1, 1.f, 0, 0);
        add_rmsnorm_kernel<<<B_ * S_, 256, 0, stream>>>(x, tmp2, n1w + (long)l * E_, x);

        // FFN: hmid = gelu(x W1^T + b1); x = rmsnorm(x + hmid W2^T + b2, n2)
        gemm(x, E_, 0, 0,
             fw1 + (long)l * FF_ * E_, E_, 0, 0,
             fb1 + (long)l * FF_, 0, 0,
             big, FF_, 0, 0,
             B_ * S_, FF_, E_, 1, 1, 1.f, 1, 0);
        gemm(big, FF_, 0, 0,
             fw2 + (long)l * E_ * FF_, FF_, 0, 0,
             fb2 + (long)l * E_, 0, 0,
             tmp2, E_, 0, 0,
             B_ * S_, E_, FF_, 1, 1, 1.f, 0, 0);
        add_rmsnorm_kernel<<<B_ * S_, 256, 0, stream>>>(x, tmp2, n2w + (long)l * E_, x);
    }

    // final norm + LM head over image tokens only -> d_out (B, T2, 768)
    add_rmsnorm_kernel<<<B_ * S_, 256, 0, stream>>>(x, nullptr, normw, tmp1);
    gemm(tmp1 + (long)T1_ * E_, E_, (long)S_ * E_, 0,
         lmw, E_, 0, 0,
         lmb, 0, 0,
         out, 3 * GRID_ * GRID_, (long)T2_ * 3 * GRID_ * GRID_, 0,
         T2_, 3 * GRID_ * GRID_, E_, B_, 1, 1.f, 0, 0);
}
